// RNNEncoder_8306466750919
// MI455X (gfx1250) — matrix operand order repaired and checked by reading, not compiled, not run on hardware
//
#include <hip/hip_runtime.h>
#include <hip/hip_bf16.h>

// ---------------------------------------------------------------------------
// 2-layer GRU (SEQ=4096, B=64, H=512) persistent-kernel implementation for
// gfx1250. Recurrent GEMMs use v_wmma_f32_16x16x32_bf16; weights staged to
// LDS once in chunk-blocked layout (48KB blocks so every ds_load offset fits
// the 16-bit immediate); K-loop fully unrolled; h state lives in registers;
// grid barrier per timestep.
// ---------------------------------------------------------------------------

typedef __attribute__((ext_vector_type(16))) __bf16 v16bf;
typedef __attribute__((ext_vector_type(8)))  float  v8f;

#define SEQ   4096
#define BATCH 64
#define HDIM  512
#define GDIM  (3 * HDIM)
#define NWG0  8            // layer-0 workgroups (64 hidden units each)
#define NWG1  16           // layer-1 workgroups (32 hidden units each)
#define NWG   (NWG0 + NWG1)
#define BLKROWS 48                       // {r,z,n} x 16 hidden units
#define BLKELTS (BLKROWS * HDIM)         // 24576 bf16 = 49152 B per block
#define LDS_BYTES (4 * BLKELTS * 2)      // 196608 B

union Frag { v16bf v; uint4 q[2]; };

__device__ __forceinline__ float sigmoidf_(float v) {
  return 1.0f / (1.0f + __expf(-v));
}

// A-matrix fragment (16x32 bf16, M=batch rows, K=hidden): lanes 0-15 hold
// row M=lane with K kk..kk+15; lanes 16-31 hold K kk+16..kk+31 (matches B).
// hbuf is row-major (BATCH x HDIM) bf16.
__device__ __forceinline__ v16bf load_A_global(const __bf16* hbuf, int mbase,
                                               int kk, int lane) {
  const int row = mbase + (lane & 15);
  const int c0  = kk + ((lane >> 4) << 4);   // +0 or +16
  const char* p = (const char*)(hbuf + (long long)row * HDIM + c0);
  Frag f;
  f.q[0] = *(const uint4*)(p);
  f.q[1] = *(const uint4*)(p + 16);          // +8 elements
  return f.v;
}

// B-matrix fragment (32x16 bf16, K x gate-col) from one 48-row LDS block.
// Block layout: rows [0,16)=r, [16,32)=z, [32,48)=n, each row = 512 K bf16.
// B[k][g] = W[g][k]: lane c<16 -> gate column c, K=kk..kk+15 (32 contiguous
// bytes); lanes 16-31 -> K=kk+16..kk+31. All offsets < 48KB -> ds immediates.
__device__ __forceinline__ v16bf load_B_blk(const __bf16* blk, int gatekind,
                                            int kk, int lane) {
  const int row = gatekind * 16 + (lane & 15);
  const int c0  = kk + ((lane >> 4) << 4);   // +0 or +16
  const uint4* p = (const uint4*)(blk + row * HDIM + c0);
  Frag f;
  f.q[0] = p[0];
  f.q[1] = p[1];
  return f.v;
}

__device__ __forceinline__ v8f wmma_bf16(v16bf a, v16bf b, v8f c) {
  return __builtin_amdgcn_wmma_f32_16x16x32_bf16(
      /*neg_a=*/false, a, /*neg_b=*/false, b,
      /*c_mod=*/(short)0, c, /*reuse_a=*/false, /*reuse_b=*/false);
}

__device__ __forceinline__ void grid_barrier(unsigned* cnt, unsigned target) {
  __syncthreads();
  if (threadIdx.x == 0) {
    __hip_atomic_fetch_add(cnt, 1u, __ATOMIC_RELEASE, __HIP_MEMORY_SCOPE_AGENT);
    while (__hip_atomic_load(cnt, __ATOMIC_ACQUIRE, __HIP_MEMORY_SCOPE_AGENT)
           < target) {
      __builtin_amdgcn_s_sleep(2);
    }
  }
  __syncthreads();
}

// ---------------------------------------------------------------------------
// Prep: convert fp32 weights -> bf16 in workspace, zero h buffers + counter.
// ---------------------------------------------------------------------------
__global__ void gru_prep(const float* __restrict__ whh0,
                         const float* __restrict__ whh1,
                         const float* __restrict__ wih1,
                         __bf16* __restrict__ wb,
                         __bf16* __restrict__ hb,
                         unsigned* __restrict__ cnt) {
  const long long NW = (long long)GDIM * HDIM;
  const long long idx = (long long)blockIdx.x * blockDim.x + threadIdx.x;
  const long long stride = (long long)gridDim.x * blockDim.x;
  for (long long i = idx; i < 3 * NW; i += stride) {
    float v;
    if (i < NW)            v = whh0[i];
    else if (i < 2 * NW)   v = whh1[i - NW];
    else                   v = wih1[i - 2 * NW];
    wb[i] = (__bf16)v;
  }
  const long long NH = 4LL * BATCH * HDIM;   // h0b[2] + h1b[2]
  for (long long i = idx; i < NH; i += stride) hb[i] = (__bf16)0.0f;
  if (idx == 0) *cnt = 0u;
}

// ---------------------------------------------------------------------------
// Persistent GRU kernel. 24 blocks x 256 threads (8 waves).
// ---------------------------------------------------------------------------
__global__ void __launch_bounds__(256, 1)
gru_persistent(const float* __restrict__ x,
               const float* __restrict__ wih0,
               const float* __restrict__ bih0,
               const float* __restrict__ bhh0,
               const float* __restrict__ bih1,
               const float* __restrict__ bhh1,
               const __bf16* __restrict__ wb_hh0,
               const __bf16* __restrict__ wb_hh1,
               const __bf16* __restrict__ wb_ih1,
               __bf16* __restrict__ h0b,     // 2 x (BATCH*HDIM)
               __bf16* __restrict__ h1b,     // 2 x (BATCH*HDIM)
               unsigned* __restrict__ cnt,
               float* __restrict__ out) {    // (2, BATCH, HDIM)
  extern __shared__ __bf16 lds[];
  const int tid  = threadIdx.x;
  const int lane = tid & 31;
  const int wv   = tid >> 5;
  const int bid  = blockIdx.x;
  const bool isL0 = (bid < NWG0);

  // ---- Stage this WG's weight slice into LDS (once), chunk-blocked. ----
  // 4 blocks of 48 rows x 512 K. Within a block: rows [0,16)=r, [16,32)=z,
  // [32,48)=n for one 16-wide hidden chunk.
  if (isL0) {
    const int Hb = bid * 64;   // owns hidden units [Hb, Hb+64)
    for (int i = tid; i < 4 * BLKROWS * (HDIM / 8); i += blockDim.x) {
      const int blkq = BLKROWS * (HDIM / 8);      // uint4s per block
      const int blk = i / blkq;
      const int rem = i - blk * blkq;
      const int r = rem >> 6, c = rem & 63;
      const int gate = (r >> 4) * HDIM + Hb + blk * 16 + (r & 15);
      ((uint4*)lds)[i] =
          ((const uint4*)(wb_hh0 + (long long)gate * HDIM))[c];
    }
  } else {
    const int Hb = (bid - NWG0) * 32;  // owns hidden units [Hb, Hb+32)
    // Blocks: 0=whh1 chunk0, 1=whh1 chunk1, 2=wih1 chunk0, 3=wih1 chunk1.
    for (int i = tid; i < 4 * BLKROWS * (HDIM / 8); i += blockDim.x) {
      const int blkq = BLKROWS * (HDIM / 8);
      const int blk = i / blkq;
      const int rem = i - blk * blkq;
      const int r = rem >> 6, c = rem & 63;
      const __bf16* base = (blk < 2) ? wb_hh1 : wb_ih1;
      const int chunk = blk & 1;
      const int gate = (r >> 4) * HDIM + Hb + chunk * 16 + (r & 15);
      ((uint4*)lds)[i] =
          ((const uint4*)(base + (long long)gate * HDIM))[c];
    }
  }
  __syncthreads();

  const int mtile = wv & 3;
  const int mbase = mtile * 16;
  const v8f vzero = {0.f, 0.f, 0.f, 0.f, 0.f, 0.f, 0.f, 0.f};

  if (isL0) {
    // ================= layer 0 =================
    const int Hb  = bid * 64;
    const int hc0 = (wv >> 2) * 2;   // two 16-wide hidden chunks
    const __bf16* blk0 = lds + (long long)(hc0)     * BLKELTS;
    const __bf16* blk1 = lds + (long long)(hc0 + 1) * BLKELTS;

    float hreg[2][8];
#pragma unroll
    for (int c = 0; c < 2; ++c)
#pragma unroll
      for (int v = 0; v < 8; ++v) hreg[c][v] = 0.0f;

    // Hoist per-lane gate constants (fixed hidden column per lane/chunk).
    float wr[2], wz[2], wn[2], brz[2], bzz[2], bin_[2], bhn_[2];
    int   jgc[2];
#pragma unroll
    for (int c = 0; c < 2; ++c) {
      const int jg = Hb + (hc0 + c) * 16 + (lane & 15);
      jgc[c]  = jg;
      wr[c]   = wih0[jg];
      wz[c]   = wih0[HDIM + jg];
      wn[c]   = wih0[2 * HDIM + jg];
      brz[c]  = bih0[jg] + bhh0[jg];
      bzz[c]  = bih0[HDIM + jg] + bhh0[HDIM + jg];
      bin_[c] = bih0[2 * HDIM + jg];
      bhn_[c] = bhh0[2 * HDIM + jg];
    }

    for (int t = 0; t <= SEQ; ++t) {
      if (t < SEQ) {
        const __bf16* hprev = h0b + (long long)((t + 1) & 1) * (BATCH * HDIM);
        v8f Cr[2], Cz[2], Cn[2];
#pragma unroll
        for (int c = 0; c < 2; ++c) { Cr[c] = vzero; Cz[c] = vzero; Cn[c] = vzero; }

#pragma unroll
        for (int kk = 0; kk < HDIM; kk += 32) {
          const v16bf a = load_A_global(hprev, mbase, kk, lane);
          Cr[0] = wmma_bf16(a, load_B_blk(blk0, 0, kk, lane), Cr[0]);
          Cz[0] = wmma_bf16(a, load_B_blk(blk0, 1, kk, lane), Cz[0]);
          Cn[0] = wmma_bf16(a, load_B_blk(blk0, 2, kk, lane), Cn[0]);
          Cr[1] = wmma_bf16(a, load_B_blk(blk1, 0, kk, lane), Cr[1]);
          Cz[1] = wmma_bf16(a, load_B_blk(blk1, 1, kk, lane), Cz[1]);
          Cn[1] = wmma_bf16(a, load_B_blk(blk1, 2, kk, lane), Cn[1]);
        }

        __bf16* hout = h0b + (long long)(t & 1) * (BATCH * HDIM);
#pragma unroll
        for (int c = 0; c < 2; ++c) {
          const int jg = jgc[c];
#pragma unroll
          for (int v = 0; v < 8; ++v) {
            const int b = mbase + ((lane >> 4) << 3) + v;
            const float xb = x[(long long)t * BATCH + b];
            const float r  = sigmoidf_(Cr[c][v] + xb * wr[c] + brz[c]);
            const float z  = sigmoidf_(Cz[c][v] + xb * wz[c] + bzz[c]);
            const float n  = tanhf((xb * wn[c] + bin_[c]) +
                                   r * (Cn[c][v] + bhn_[c]));
            const float hn = (1.0f - z) * n + z * hreg[c][v];
            hreg[c][v] = hn;
            hout[(long long)b * HDIM + jg] = (__bf16)hn;
            if (t == SEQ - 1) out[(long long)b * HDIM + jg] = hn;
          }
        }
      }
      grid_barrier(cnt, (unsigned)(t + 1) * NWG);
    }
  } else {
    // ================= layer 1 (skewed by one step) =================
    const int Hb = (bid - NWG0) * 32;
    const int hc = (wv >> 2);          // one 16-wide hidden chunk
    const __bf16* blkh = lds + (long long)(hc)     * BLKELTS;  // whh1 chunk
    const __bf16* blkx = lds + (long long)(2 + hc) * BLKELTS;  // wih1 chunk

    float hreg[8];
#pragma unroll
    for (int v = 0; v < 8; ++v) hreg[v] = 0.0f;

    const int jg = Hb + hc * 16 + (lane & 15);
    const float brz  = bih1[jg] + bhh1[jg];
    const float bzz  = bih1[HDIM + jg] + bhh1[HDIM + jg];
    const float bin_ = bih1[2 * HDIM + jg];
    const float bhn_ = bhh1[2 * HDIM + jg];

    for (int t = 0; t <= SEQ; ++t) {
      if (t >= 1) {
        const int s = t - 1;
        const __bf16* h1prev = h1b + (long long)((s + 1) & 1) * (BATCH * HDIM);
        const __bf16* h0cur  = h0b + (long long)(s & 1) * (BATCH * HDIM);
        v8f Cr = vzero, Cz = vzero, Cn = vzero;   // recurrent (h1 @ Whh1^T)
        v8f Dr = vzero, Dz = vzero, Dn = vzero;   // input     (h0 @ Wih1^T)

#pragma unroll
        for (int kk = 0; kk < HDIM; kk += 32) {
          const v16bf ah = load_A_global(h1prev, mbase, kk, lane);
          const v16bf ax = load_A_global(h0cur, mbase, kk, lane);
          Cr = wmma_bf16(ah, load_B_blk(blkh, 0, kk, lane), Cr);
          Cz = wmma_bf16(ah, load_B_blk(blkh, 1, kk, lane), Cz);
          Cn = wmma_bf16(ah, load_B_blk(blkh, 2, kk, lane), Cn);
          Dr = wmma_bf16(ax, load_B_blk(blkx, 0, kk, lane), Dr);
          Dz = wmma_bf16(ax, load_B_blk(blkx, 1, kk, lane), Dz);
          Dn = wmma_bf16(ax, load_B_blk(blkx, 2, kk, lane), Dn);
        }

        __bf16* hout = h1b + (long long)(s & 1) * (BATCH * HDIM);
#pragma unroll
        for (int v = 0; v < 8; ++v) {
          const int b = mbase + ((lane >> 4) << 3) + v;
          const float r  = sigmoidf_(Cr[v] + Dr[v] + brz);
          const float z  = sigmoidf_(Cz[v] + Dz[v] + bzz);
          const float n  = tanhf((Dn[v] + bin_) + r * (Cn[v] + bhn_));
          const float hn = (1.0f - z) * n + z * hreg[v];
          hreg[v] = hn;
          hout[(long long)b * HDIM + jg] = (__bf16)hn;
          if (s == SEQ - 1)
            out[(long long)(BATCH * HDIM) + (long long)b * HDIM + jg] = hn;
        }
      }
      grid_barrier(cnt, (unsigned)(t + 1) * NWG);
    }
  }
}

// ---------------------------------------------------------------------------
extern "C" void kernel_launch(void* const* d_in, const int* in_sizes, int n_in,
                              void* d_out, int out_size, void* d_ws,
                              size_t ws_size, hipStream_t stream) {
  (void)in_sizes; (void)n_in; (void)out_size; (void)ws_size;

  const float* x    = (const float*)d_in[0];
  const float* wih0 = (const float*)d_in[1];
  const float* whh0 = (const float*)d_in[2];
  const float* bih0 = (const float*)d_in[3];
  const float* bhh0 = (const float*)d_in[4];
  const float* wih1 = (const float*)d_in[5];
  const float* whh1 = (const float*)d_in[6];
  const float* bih1 = (const float*)d_in[7];
  const float* bhh1 = (const float*)d_in[8];
  float* out = (float*)d_out;

  // Workspace layout (all 16B-aligned):
  //   [0,256)                          : barrier counter
  //   [256, 256+3*G*H*2)               : bf16 weights {whh0, whh1, wih1}
  //   then 2x(B*H) bf16 h0 buffers, 2x(B*H) bf16 h1 buffers.
  char* ws = (char*)d_ws;
  unsigned* cnt = (unsigned*)ws;
  __bf16* wb   = (__bf16*)(ws + 256);
  const long long WBYTES = 3LL * GDIM * HDIM * 2;
  __bf16* h0bp = (__bf16*)(ws + 256 + WBYTES);
  __bf16* h1bp = h0bp + 2LL * BATCH * HDIM;

  gru_prep<<<1024, 256, 0, stream>>>(whh0, whh1, wih1, wb, h0bp, cnt);

  static int attr_done = 0;  // setting an attribute is idempotent/deterministic
  if (!attr_done) {
    (void)hipFuncSetAttribute((const void*)gru_persistent,
                              hipFuncAttributeMaxDynamicSharedMemorySize,
                              LDS_BYTES);
    attr_done = 1;
  }

  const __bf16* wb_hh0 = wb;
  const __bf16* wb_hh1 = wb + (long long)GDIM * HDIM;
  const __bf16* wb_ih1 = wb + 2LL * GDIM * HDIM;

  gru_persistent<<<NWG, 256, LDS_BYTES, stream>>>(
      x, wih0, bih0, bhh0, bih1, bhh1,
      wb_hh0, wb_hh1, wb_ih1, h0bp, h1bp, cnt, out);
}
